// QuantizedLinear_23519240912993
// MI455X (gfx1250) — matrix operand order repaired and checked by reading, not compiled, not run on hardware
//
#include <hip/hip_runtime.h>

typedef __attribute__((ext_vector_type(16))) _Float16 v16h;
typedef __attribute__((ext_vector_type(8)))  _Float16 v8h;
typedef __attribute__((ext_vector_type(8)))  float    v8f;
typedef __attribute__((ext_vector_type(4)))  float    v4f;
typedef __attribute__((ext_vector_type(4)))  int      v4i;

#define TOKENS 2048
#define OUTF   11008
#define INFEAT 4096
#define BM 128
#define BN 256
#define BK 64
#define LDK 72            // padded LDS K-stride in halves (144 B rows)
#define NSC (INFEAT/128)  // 32 scales per output row
#define NSLAB (INFEAT/BK) // 64 K-slabs

union FragU { v16h v; v8h h[2]; };

__global__ __launch_bounds__(256) void qlinear_wmma_f16(
    const float* __restrict__ X,
    const int* __restrict__ Wc,
    const _Float16* __restrict__ Wsc,
    const float* __restrict__ Bias,
    float* __restrict__ Out)
{
  // ping-pong buffers: 2 * (128+256) rows * 144 B = 110,592 B of LDS
  __shared__ __align__(16) _Float16 XsBuf[2][BM][LDK];
  __shared__ __align__(16) _Float16 WsBuf[2][BN][LDK];

  const int t    = threadIdx.x;
  const int lane = t & 31;
  const int wave = t >> 5;
  const int wm   = wave >> 2;   // 0..1 : M 2x64
  const int wn   = wave & 3;    // 0..3 : N 4x64
  const int mBlock = blockIdx.y * BM;
  const int nBlock = blockIdx.x * BN;

  // X staging: 2 threads per row (32 floats each); W staging: 1 thread per row (64 codes)
  const int xrow = t >> 1;
  const int xkc  = (t & 1) * 32;
  const int wrow = t;

  const float*    xsrc = X   + (size_t)(mBlock + xrow) * INFEAT + xkc;
  const int*      wsrc = Wc  + (size_t)(nBlock + wrow) * INFEAT;
  const _Float16* ssrc = Wsc + (size_t)(nBlock + wrow) * NSC;

  v8h hx[4];   // staged X slice, f16
  v8h hw[8];   // staged W slice, dequantized f16

  // ---- load + convert one K-slab into registers ----
  auto stage = [&](int k0) {
#pragma unroll
    for (int i = 0; i < 4; ++i) {
      v4f a = *(const v4f*)(xsrc + k0 + i * 8);
      v4f b = *(const v4f*)(xsrc + k0 + i * 8 + 4);
      v8h h;
      h[0] = (_Float16)a[0]; h[1] = (_Float16)a[1];
      h[2] = (_Float16)a[2]; h[3] = (_Float16)a[3];
      h[4] = (_Float16)b[0]; h[5] = (_Float16)b[1];
      h[6] = (_Float16)b[2]; h[7] = (_Float16)b[3];
      hx[i] = h;
    }
    const _Float16 sc = ssrc[k0 >> 7];   // [k0,k0+64) stays in one 128-block
#pragma unroll
    for (int i = 0; i < 8; ++i) {
      v4i a = __builtin_nontemporal_load((const v4i*)(wsrc + k0 + i * 8));
      v4i b = __builtin_nontemporal_load((const v4i*)(wsrc + k0 + i * 8 + 4));
      v8h h;   // i16 -> f16 convert, then packed f16 multiply by scale
      h[0] = (_Float16)(short)a[0] * sc;
      h[1] = (_Float16)(short)a[1] * sc;
      h[2] = (_Float16)(short)a[2] * sc;
      h[3] = (_Float16)(short)a[3] * sc;
      h[4] = (_Float16)(short)b[0] * sc;
      h[5] = (_Float16)(short)b[1] * sc;
      h[6] = (_Float16)(short)b[2] * sc;
      h[7] = (_Float16)(short)b[3] * sc;
      hw[i] = h;
    }
  };

  // ---- commit staged registers to LDS buffer `buf` ----
  auto commit = [&](int buf) {
    _Float16* xd = &XsBuf[buf][xrow][xkc];
#pragma unroll
    for (int i = 0; i < 4; ++i) *(v8h*)(xd + i * 8) = hx[i];
    _Float16* wd = &WsBuf[buf][wrow][0];
#pragma unroll
    for (int i = 0; i < 8; ++i) *(v8h*)(wd + i * 8) = hw[i];
  };

  v8f acc[4][4];
#pragma unroll
  for (int tm = 0; tm < 4; ++tm)
#pragma unroll
    for (int tn = 0; tn < 4; ++tn)
      acc[tm][tn] = (v8f){0.f, 0.f, 0.f, 0.f, 0.f, 0.f, 0.f, 0.f};

  // ---- prologue: fill buf0, prefetch slab 1 ----
  stage(0);
  commit(0);
  stage(BK);

  for (int i = 0; i < NSLAB; ++i) {
    const int k0 = i * BK;
    __syncthreads();  // buf(i&1) fully committed; buf((i+1)&1) free to overwrite

    if (i + 1 < NSLAB) commit((i + 1) & 1);      // overlaps with compute below
    if (i + 2 < NSLAB) stage(k0 + 2 * BK);       // global prefetch + convert

    const _Float16 (*Xs)[LDK] = XsBuf[i & 1];
    const _Float16 (*Ws)[LDK] = WsBuf[i & 1];

#pragma unroll
    for (int ks = 0; ks < BK; ks += 32) {
      const int koffB = ks + ((lane >> 4) << 3);   // lanes 16-31: K+8 start
      v16h bfr[4];
#pragma unroll
      for (int tn = 0; tn < 4; ++tn) {
        const _Float16* p = &Ws[wn * 64 + tn * 16 + (lane & 15)][koffB];
        FragU u;
        u.h[0] = *(const v8h*)p;
        u.h[1] = *(const v8h*)(p + 16);
        bfr[tn] = u.v;
      }
      const int koffA = ks + ((lane >> 4) << 3);   // lanes 16-31: K+8 start
#pragma unroll
      for (int tm = 0; tm < 4; ++tm) {
        const _Float16* p = &Xs[wm * 64 + tm * 16 + (lane & 15)][koffA];
        FragU u;
        u.h[0] = *(const v8h*)p;          // K +0..7   (or +8..15)
        u.h[1] = *(const v8h*)(p + 16);   // K +16..23 (or +24..31)
        const v16h afr = u.v;
#pragma unroll
        for (int tn = 0; tn < 4; ++tn)
          acc[tm][tn] = __builtin_amdgcn_wmma_f32_16x16x32_f16(
              false, afr, false, bfr[tn],
              (short)0, acc[tm][tn], false, false);
      }
    }
  }

  // ---- epilogue: bias add + store ----
  const int lhi  = lane >> 4;
  const int lcol = lane & 15;
  float bv[4];
#pragma unroll
  for (int tn = 0; tn < 4; ++tn)
    bv[tn] = Bias[nBlock + wn * 64 + tn * 16 + lcol];
#pragma unroll
  for (int tm = 0; tm < 4; ++tm) {
    const int r0 = mBlock + wm * 64 + tm * 16 + lhi * 8;
#pragma unroll
    for (int tn = 0; tn < 4; ++tn) {
      const int col = nBlock + wn * 64 + tn * 16 + lcol;
#pragma unroll
      for (int i = 0; i < 8; ++i)
        Out[(size_t)(r0 + i) * OUTF + col] = acc[tm][tn][i] + bv[tn];
    }
  }
}

extern "C" void kernel_launch(void* const* d_in, const int* in_sizes, int n_in,
                              void* d_out, int out_size, void* d_ws, size_t ws_size,
                              hipStream_t stream) {
  (void)in_sizes; (void)n_in; (void)out_size; (void)d_ws; (void)ws_size;
  const float*    x      = (const float*)d_in[0];
  const int*      codes  = (const int*)d_in[1];
  const _Float16* scales = (const _Float16*)d_in[2];
  const float*    bias   = (const float*)d_in[3];
  float*          out    = (float*)d_out;

  dim3 grid(OUTF / BN, TOKENS / BM);   // 43 x 16
  qlinear_wmma_f16<<<grid, 256, 0, stream>>>(x, codes, scales, bias, out);
}
